// GU_9320079033011
// MI455X (gfx1250) — matrix operand order repaired and checked by reading, not compiled, not run on hardware
//
#include <hip/hip_runtime.h>

// ---------------------------------------------------------------------------
// 2-layer GRU for MI455X (gfx1250, wave32, WMMA, async-to-LDS).
//   Phase A: one-time f32->bf16 converts + weight transposes (K-major).
//   Phase B: WMMA GEMM for input projections (double-buffered LDS, ASYNCcnt).
//   Phase C: persistent recurrent scan, Whh slice resident in 320KB WGP LDS,
//            grid barrier per timestep, async h broadcast, xg prefetch.
// ---------------------------------------------------------------------------

#define B_   32
#define T_   512
#define D_   512
#define H_   1024
#define G3_  3072   // 3*H

typedef __bf16 bf16_t;
typedef __bf16 bf16x16 __attribute__((ext_vector_type(16)));
typedef float  f32x8   __attribute__((ext_vector_type(8)));
typedef int    i32x4   __attribute__((ext_vector_type(4)));

// ---- CDNA5 async global->LDS path (guarded so the build never regresses) --
#if defined(__has_builtin)
#if __has_builtin(__builtin_amdgcn_global_load_async_to_lds_b128)
#define HAVE_ASYNC_LDS 1
#endif
#if __has_builtin(__builtin_amdgcn_s_wait_asynccnt)
#define HAVE_WAIT_ASYNC 1
#endif
#endif
#ifndef HAVE_ASYNC_LDS
#define HAVE_ASYNC_LDS 0
#endif
#ifndef HAVE_WAIT_ASYNC
#define HAVE_WAIT_ASYNC 0
#endif

#if HAVE_ASYNC_LDS
#if HAVE_WAIT_ASYNC
#define WAIT_ASYNC(n) __builtin_amdgcn_s_wait_asynccnt(n)
#else
#define WAIT_ASYNC(n) asm volatile("s_wait_asynccnt " #n ::: "memory")
#endif
typedef __attribute__((address_space(1))) i32x4* gbl_b128_t;
typedef __attribute__((address_space(3))) i32x4* lds_b128_t;
// 16B global -> LDS, no VGPR round-trip, tracked by ASYNCcnt.
__device__ __forceinline__ void async_b128(void* lds, const void* g) {
  __builtin_amdgcn_global_load_async_to_lds_b128(
      (gbl_b128_t)(g), (lds_b128_t)(lds), 0, 0);
}
#endif

union Frag { bf16x16 v; uint4 u[2]; };

// A-fragment (16x32 bf16, row-major M x K source, per ISA 7.12.2):
// lanes 0-15 hold M=lane, K 0..7 & 16..23; lanes 16-31 hold K 8..15 & 24..31.
__device__ __forceinline__ bf16x16 load_frag_a(const bf16_t* base, int ld) {
  int lane = threadIdx.x & 31;
  int m    = lane & 15;
  int kh   = (lane >> 4) ? 8 : 0;
  const bf16_t* p = base + m * ld + kh;
  Frag f;
  f.u[0] = *reinterpret_cast<const uint4*>(p);
  f.u[1] = *reinterpret_cast<const uint4*>(p + 16);
  return f.v;
}

// B-fragment (32x16 bf16, row-major K x N source): lane holds N = lane&15,
// lanes 0-15 K 0..7 & 16..23; lanes 16-31 K 8..15 & 24..31 (matches A).
__device__ __forceinline__ bf16x16 load_frag_b(const bf16_t* base, int ld) {
  int lane = threadIdx.x & 31;
  int n    = lane & 15;
  int kh   = (lane >> 4) ? 8 : 0;
  const bf16_t* p = base + kh * ld + n;
  bf16x16 v;
#pragma unroll
  for (int s = 0; s < 8; ++s) {
    v[s]     = p[s * ld];
    v[s + 8] = p[(s + 16) * ld];
  }
  return v;
}

__device__ __forceinline__ f32x8 wmma_bf16(bf16x16 a, bf16x16 b, f32x8 c) {
  return __builtin_amdgcn_wmma_f32_16x16x32_bf16(false, a, false, b,
                                                 (short)0, c, false, false);
}

// C/D 16x16 f32 layout: VGPR r -> M = (lane>=16 ? 8 : 0)+r, N = lane&15.
__device__ __forceinline__ void store_frag_c(float* base, int ld, f32x8 c) {
  int lane = threadIdx.x & 31;
  int n  = lane & 15;
  int mb = (lane >> 4) * 8;
#pragma unroll
  for (int r = 0; r < 8; ++r) base[(mb + r) * ld + n] = c[r];
}

// ---------------------------------------------------------------------------
__global__ void cvt_f32_bf16(const float* __restrict__ in,
                             bf16_t* __restrict__ out, int n) {
  int i = blockIdx.x * blockDim.x + threadIdx.x;
  int stride = gridDim.x * blockDim.x;
  for (; i < n; i += stride) out[i] = (bf16_t)in[i];
}

// Transpose + convert: W [N,K] f32  ->  Wt [K,N] bf16
__global__ __launch_bounds__(256) void transpose_cvt(
    const float* __restrict__ W, bf16_t* __restrict__ Wt, int N, int K) {
  __shared__ float tile[32][33];
  int n0 = blockIdx.x * 32, k0 = blockIdx.y * 32;
  int tid = threadIdx.x;
  for (int i = tid; i < 1024; i += 256) {
    int r = i >> 5, c = i & 31;
    tile[r][c] = W[(size_t)(n0 + r) * K + k0 + c];
  }
  __syncthreads();
  for (int i = tid; i < 1024; i += 256) {
    int r = i >> 5, c = i & 31;
    Wt[(size_t)(k0 + r) * N + n0 + c] = (bf16_t)tile[c][r];
  }
}

// ---------------------------------------------------------------------------
// Input-projection GEMM: Y[M,3072] = bf16( X[M,K] @ Wt[K,3072] + bias )
// Block tile 64x128, K-step 64, double-buffered LDS staged via async-to-LDS.
#define BKG  64
#define LDAg 72    // 64x64 A tile pitch (144B rows, 16B aligned)
#define LDBg 136   // 64x128 B tile pitch (272B rows, 16B aligned)

__global__ __launch_bounds__(256) void gemm_xproj(
    const bf16_t* __restrict__ X, const bf16_t* __restrict__ Wt,
    const float* __restrict__ bias, bf16_t* __restrict__ Y, int M, int K) {
  __shared__ bf16_t As[2][64 * LDAg];
  __shared__ bf16_t Bs[2][BKG * LDBg];
  const int tid  = threadIdx.x;
  const int wave = tid >> 5;
  const int m0 = blockIdx.y * 64;
  const int n0 = blockIdx.x * 128;
  const int wm = (wave >> 2) * 32;
  const int wn = (wave & 3) * 32;
  const int srow = tid >> 2;          // staging row (64 rows, 4 threads each)
  const int sa   = (tid & 3) * 16;    // A: 32B per thread
  const int sb   = (tid & 3) * 32;    // B: 64B per thread
  f32x8 acc[2][2] = {};

  auto stage = [&](int buf, int k0) {
    const bf16_t* ga = &X[(size_t)(m0 + srow) * K + k0 + sa];
    bf16_t* la = &As[buf][srow * LDAg + sa];
    const bf16_t* gb = &Wt[(size_t)(k0 + srow) * G3_ + n0 + sb];
    bf16_t* lb = &Bs[buf][srow * LDBg + sb];
#if HAVE_ASYNC_LDS
    async_b128(la, ga);           async_b128(la + 8, ga + 8);
    async_b128(lb, gb);           async_b128(lb + 8, gb + 8);
    async_b128(lb + 16, gb + 16); async_b128(lb + 24, gb + 24);
#else
    reinterpret_cast<uint4*>(la)[0] = reinterpret_cast<const uint4*>(ga)[0];
    reinterpret_cast<uint4*>(la)[1] = reinterpret_cast<const uint4*>(ga)[1];
    reinterpret_cast<uint4*>(lb)[0] = reinterpret_cast<const uint4*>(gb)[0];
    reinterpret_cast<uint4*>(lb)[1] = reinterpret_cast<const uint4*>(gb)[1];
    reinterpret_cast<uint4*>(lb)[2] = reinterpret_cast<const uint4*>(gb)[2];
    reinterpret_cast<uint4*>(lb)[3] = reinterpret_cast<const uint4*>(gb)[3];
#endif
  };

  stage(0, 0);
  const int nk = K / BKG;
  for (int ki = 0; ki < nk; ++ki) {
    const int buf = ki & 1;
    if (ki + 1 < nk) {
      stage(buf ^ 1, (ki + 1) * BKG);  // overlaps with this iter's WMMAs
#if HAVE_ASYNC_LDS
      WAIT_ASYNC(6);                   // 6 in flight = next stage only
#endif
    } else {
#if HAVE_ASYNC_LDS
      WAIT_ASYNC(0);
#endif
    }
    __syncthreads();
#pragma unroll
    for (int kc = 0; kc < BKG; kc += 32) {
      bf16x16 a0 = load_frag_a(&As[buf][(wm + 0) * LDAg + kc], LDAg);
      bf16x16 a1 = load_frag_a(&As[buf][(wm + 16) * LDAg + kc], LDAg);
      bf16x16 b0 = load_frag_b(&Bs[buf][kc * LDBg + wn + 0], LDBg);
      bf16x16 b1 = load_frag_b(&Bs[buf][kc * LDBg + wn + 16], LDBg);
      acc[0][0] = wmma_bf16(a0, b0, acc[0][0]);
      acc[0][1] = wmma_bf16(a0, b1, acc[0][1]);
      acc[1][0] = wmma_bf16(a1, b0, acc[1][0]);
      acc[1][1] = wmma_bf16(a1, b1, acc[1][1]);
    }
    __syncthreads();
  }

  const int lane = tid & 31;
  const int n_in = lane & 15;
  const int mb   = (lane >> 4) * 8;
#pragma unroll
  for (int mt = 0; mt < 2; ++mt)
#pragma unroll
    for (int nt = 0; nt < 2; ++nt) {
      int gn = n0 + wn + nt * 16 + n_in;
      float bv = bias[gn];
#pragma unroll
      for (int r = 0; r < 8; ++r) {
        int gm = m0 + wm + mt * 16 + mb + r;
        Y[(size_t)gm * G3_ + gn] = (bf16_t)(acc[mt][nt][r] + bv);
      }
    }
}

// ---------------------------------------------------------------------------
// Persistent recurrent scan. 32 blocks x 192 threads (6 waves).
// Block b owns hidden columns [32b, 32b+32). Whh slice (1024x96 bf16) is LDS-
// resident for the whole kernel (208KB of the 320KB WGP LDS); h broadcast via
// global bf16 buffer + grid barrier each step, staged with async-to-LDS.
#define SCAN_BLOCKS  32
#define SCAN_THREADS 192
#define WLD 104    // weight LDS row pitch (bf16), 208B rows
#define HLD 1032   // h LDS row pitch (bf16), 2064B rows
#define GLD 100    // hg spill pitch (f32)

__global__ __launch_bounds__(SCAN_THREADS) void gru_scan(
    const bf16_t* __restrict__ Wt_hh,   // [H_, G3_] K-major
    const bf16_t* __restrict__ xg,      // [B_*T_, G3_] incl. bih
    const float* __restrict__ bhh,      // [G3_]
    bf16_t* __restrict__ hbuf,          // [B_, H_] bf16 running hidden
    bf16_t* __restrict__ out_bf16,      // [B_*T_, H_] or nullptr
    float* __restrict__ out_f32,        // [B_*T_, H_] or nullptr
    float* __restrict__ state_out,      // [B_, H_] final hidden (f32)
    unsigned* __restrict__ barrier_cnt) {
  extern __shared__ char smem[];
  bf16_t* wLDS = (bf16_t*)smem;                              // 1024 x WLD
  bf16_t* hLDS = (bf16_t*)(smem + (size_t)H_ * WLD * 2);     // 32 x HLD
  float* hgbuf = (float*)(smem + (size_t)H_ * WLD * 2 + (size_t)B_ * HLD * 2);
  float* bhhL  = (float*)((char*)hgbuf + (size_t)B_ * GLD * 4);  // [96]

  const int tid  = threadIdx.x;
  const int wave = tid >> 5;
  const int c0   = blockIdx.x * 32;

  // Resident weight slice: Wt_hh[:, g*H + c0 .. c0+31] for g = 0..2
  for (int r = tid; r < H_; r += SCAN_THREADS) {
#pragma unroll
    for (int g = 0; g < 3; ++g) {
      const bf16_t* src = &Wt_hh[(size_t)r * G3_ + g * H_ + c0];
      bf16_t* dst = &wLDS[r * WLD + g * 32];
#if HAVE_ASYNC_LDS
      async_b128(dst, src);           async_b128(dst + 8, src + 8);
      async_b128(dst + 16, src + 16); async_b128(dst + 24, src + 24);
#else
      reinterpret_cast<uint4*>(dst)[0] = reinterpret_cast<const uint4*>(src)[0];
      reinterpret_cast<uint4*>(dst)[1] = reinterpret_cast<const uint4*>(src)[1];
      reinterpret_cast<uint4*>(dst)[2] = reinterpret_cast<const uint4*>(src)[2];
      reinterpret_cast<uint4*>(dst)[3] = reinterpret_cast<const uint4*>(src)[3];
#endif
    }
  }
  for (int i = tid; i < 96; i += SCAN_THREADS) {
    int g = i >> 5, j = i & 31;
    bhhL[i] = bhh[g * H_ + c0 + j];
  }
  auto load_h = [&]() {  // broadcast full h (32x1024 bf16) into LDS
    for (int i = tid; i < (B_ * H_) / 8; i += SCAN_THREADS) {
      int row = i >> 7, ch = i & 127;
      bf16_t* dst = &hLDS[row * HLD + ch * 8];
      const bf16_t* src = &hbuf[row * H_ + ch * 8];
#if HAVE_ASYNC_LDS
      async_b128(dst, src);
#else
      *reinterpret_cast<uint4*>(dst) = *reinterpret_cast<const uint4*>(src);
#endif
    }
  };
  load_h();
#if HAVE_ASYNC_LDS
  WAIT_ASYNC(0);
#endif
  __syncthreads();

  const int g  = wave >> 1;  // gate 0..2
  const int nt = wave & 1;   // 16-col half of our 32 columns

  for (int t = 0; t < T_; ++t) {
    // Prefetch this step's xg rows (96 x 64B) while the WMMA loop runs.
    for (int i = tid; i < 96; i += SCAN_THREADS) {
      int pg = i >> 5, pb = i & 31;
      __builtin_prefetch(&xg[(size_t)(pb * T_ + t) * G3_ + pg * H_ + c0], 0, 1);
    }
    // hg tile (32 x 16 per wave, gate g): K = 1024 in 32-chunks
    f32x8 acc0 = {}, acc1 = {};
    for (int k0 = 0; k0 < H_; k0 += 32) {
      bf16x16 b  = load_frag_b(&wLDS[(size_t)k0 * WLD + g * 32 + nt * 16], WLD);
      bf16x16 a0 = load_frag_a(&hLDS[0 * HLD + k0], HLD);
      bf16x16 a1 = load_frag_a(&hLDS[16 * HLD + k0], HLD);
      acc0 = wmma_bf16(a0, b, acc0);
      acc1 = wmma_bf16(a1, b, acc1);
    }
    store_frag_c(&hgbuf[0 * GLD + g * 32 + nt * 16], GLD, acc0);
    store_frag_c(&hgbuf[16 * GLD + g * 32 + nt * 16], GLD, acc1);
    __syncthreads();

    // Fused gate math over our 32x32 outputs
    for (int i = tid; i < B_ * 32; i += SCAN_THREADS) {
      int b_ = i >> 5, j = i & 31;
      size_t xrow = (size_t)(b_ * T_ + t) * G3_;
      float xr = (float)xg[xrow + 0 * H_ + c0 + j];
      float xz = (float)xg[xrow + 1 * H_ + c0 + j];
      float xn = (float)xg[xrow + 2 * H_ + c0 + j];
      float hr = hgbuf[b_ * GLD + 0  + j] + bhhL[j];
      float hz = hgbuf[b_ * GLD + 32 + j] + bhhL[32 + j];
      float hn = hgbuf[b_ * GLD + 64 + j] + bhhL[64 + j];
      float r = 1.f / (1.f + __expf(-(xr + hr)));
      float z = 1.f / (1.f + __expf(-(xz + hz)));
      float pre = xn + r * hn;
      float e2  = __expf(-2.f * pre);
      float n   = (1.f - e2) / (1.f + e2);
      float hold = (float)hLDS[b_ * HLD + c0 + j];
      float hnew = (1.f - z) * n + z * hold;
      hbuf[b_ * H_ + c0 + j] = (bf16_t)hnew;
      size_t orow = (size_t)(b_ * T_ + t) * H_ + c0 + j;
      if (out_bf16) out_bf16[orow] = (bf16_t)hnew;
      if (out_f32)  out_f32[orow]  = hnew;
      if (t == T_ - 1) state_out[b_ * H_ + c0 + j] = hnew;
    }

    if (t < T_ - 1) {
      __threadfence();   // release our h slice
      __syncthreads();
      if (tid == 0) {
        __hip_atomic_fetch_add(barrier_cnt, 1u, __ATOMIC_ACQ_REL,
                               __HIP_MEMORY_SCOPE_AGENT);
        unsigned target = (unsigned)(t + 1) * SCAN_BLOCKS;
        while (__hip_atomic_load(barrier_cnt, __ATOMIC_ACQUIRE,
                                 __HIP_MEMORY_SCOPE_AGENT) < target)
          __builtin_amdgcn_s_sleep(2);
      }
      __syncthreads();
      __threadfence();   // acquire other blocks' h slices
      load_h();
#if HAVE_ASYNC_LDS
      WAIT_ASYNC(0);
#endif
      __syncthreads();
    }
  }
}

// ---------------------------------------------------------------------------
extern "C" void kernel_launch(void* const* d_in, const int* in_sizes, int n_in,
                              void* d_out, int out_size, void* d_ws,
                              size_t ws_size, hipStream_t stream) {
  (void)in_sizes; (void)n_in; (void)out_size; (void)ws_size;
  const float* x      = (const float*)d_in[0];
  const float* states = (const float*)d_in[1];
  const float* Wih0 = (const float*)d_in[2];
  const float* Whh0 = (const float*)d_in[3];
  const float* bih0 = (const float*)d_in[4];
  const float* bhh0 = (const float*)d_in[5];
  const float* Wih1 = (const float*)d_in[6];
  const float* Whh1 = (const float*)d_in[7];
  const float* bih1 = (const float*)d_in[8];
  const float* bhh1 = (const float*)d_in[9];
  float* out = (float*)d_out;

  // Workspace carve-up (≈175 MB)
  char* ws = (char*)d_ws;
  size_t off = 0;
  auto alloc = [&](size_t bytes) {
    char* p = ws + off;
    off = (off + bytes + 255) & ~(size_t)255;
    return p;
  };
  unsigned* counters = (unsigned*)alloc(256);                     // 2 barriers
  bf16_t* x0b   = (bf16_t*)alloc((size_t)B_ * T_ * D_ * 2);
  bf16_t* wtih0 = (bf16_t*)alloc((size_t)D_ * G3_ * 2);
  bf16_t* wthh0 = (bf16_t*)alloc((size_t)H_ * G3_ * 2);
  bf16_t* wtih1 = (bf16_t*)alloc((size_t)H_ * G3_ * 2);
  bf16_t* wthh1 = (bf16_t*)alloc((size_t)H_ * G3_ * 2);
  bf16_t* xgbuf = (bf16_t*)alloc((size_t)B_ * T_ * G3_ * 2);
  bf16_t* out0b = (bf16_t*)alloc((size_t)B_ * T_ * H_ * 2);
  bf16_t* h0b   = (bf16_t*)alloc((size_t)B_ * H_ * 2);
  bf16_t* h1b   = (bf16_t*)alloc((size_t)B_ * H_ * 2);

  (void)hipMemsetAsync(counters, 0, 256, stream);

  // Phase A: converts + transposes
  cvt_f32_bf16<<<2048, 256, 0, stream>>>(x, x0b, B_ * T_ * D_);
  cvt_f32_bf16<<<128, 256, 0, stream>>>(states, h0b, B_ * H_);
  cvt_f32_bf16<<<128, 256, 0, stream>>>(states + (size_t)B_ * H_, h1b, B_ * H_);
  transpose_cvt<<<dim3(G3_ / 32, D_ / 32), 256, 0, stream>>>(Wih0, wtih0, G3_, D_);
  transpose_cvt<<<dim3(G3_ / 32, H_ / 32), 256, 0, stream>>>(Whh0, wthh0, G3_, H_);
  transpose_cvt<<<dim3(G3_ / 32, H_ / 32), 256, 0, stream>>>(Wih1, wtih1, G3_, H_);
  transpose_cvt<<<dim3(G3_ / 32, H_ / 32), 256, 0, stream>>>(Whh1, wthh1, G3_, H_);

  const size_t scan_smem = (size_t)H_ * WLD * 2 + (size_t)B_ * HLD * 2 +
                           (size_t)B_ * GLD * 4 + 96 * 4;  // 292,224 B < 320 KB
  float* stateOutBase = out + (size_t)B_ * T_ * H_;

  // Layer 0
  gemm_xproj<<<dim3(G3_ / 128, (B_ * T_) / 64), 256, 0, stream>>>(
      x0b, wtih0, bih0, xgbuf, B_ * T_, D_);
  gru_scan<<<SCAN_BLOCKS, SCAN_THREADS, scan_smem, stream>>>(
      wthh0, xgbuf, bhh0, h0b, out0b, nullptr, stateOutBase, counters);

  // Layer 1
  gemm_xproj<<<dim3(G3_ / 128, (B_ * T_) / 64), 256, 0, stream>>>(
      out0b, wtih1, bih1, xgbuf, B_ * T_, H_);
  gru_scan<<<SCAN_BLOCKS, SCAN_THREADS, scan_smem, stream>>>(
      wthh1, xgbuf, bhh1, h1b, nullptr, out,
      stateOutBase + (size_t)B_ * H_, counters + 32);
}
